// TemporalTransformerBlock_20899310863001
// MI455X (gfx1250) — compile-verified, matrix operand order checked by reading, not run on hardware
//
#include <hip/hip_runtime.h>

// ---------------------------------------------------------------------------
// Fused temporal-transformer kernel for MI455X (gfx1250, wave32, WMMA + TDM).
// One workgroup (128 threads / 4 waves) processes 8 sequences (192 rows x 16).
// Dense GEMMs -> v_wmma_f32_16x16x32_f16 ; attention (head_dim=4) -> VALU;
// input tile staged via tensor_load_to_lds (TDM, 6-arg builtin).
// ---------------------------------------------------------------------------

typedef __attribute__((ext_vector_type(16))) _Float16 v16h;
typedef __attribute__((ext_vector_type(4)))  _Float16 h4v;
typedef __attribute__((ext_vector_type(8)))  float    v8f;
typedef __attribute__((ext_vector_type(4)))  unsigned int v4u;
typedef __attribute__((ext_vector_type(8)))  int      v8i;
typedef __attribute__((ext_vector_type(4)))  int      v4i;

#define D_MODEL   16
#define T_SEQ     24
#define NHEAD     4
#define HEAD_DIM  4
#define DFF       64
#define NUM_LAYERS 2
#define LN_EPS    1e-5f

#define SEQ_PER_WG 8
#define ROWS       (SEQ_PER_WG * T_SEQ)   // 192
#define MTILES     (ROWS / 16)            // 12
#define NTHREADS   128
#define NWAVES     4

// LDS partition (bytes)
#define HF_OFF 0                          // float hf[192][16]   activations
#define CF_OFF (HF_OFF + ROWS*16*4)       // float cf[192][16]   attn context
#define SF_OFF (CF_OFF + ROWS*16*4)       // float sf[192][64]   qkv / ff scratch
#define AH_OFF (SF_OFF + ROWS*64*4)       // half  ah[192][64]   f16 A staging
#define XS_OFF (AH_OFF + ROWS*64*2)       // float xs[24][8]     TDM input tile
#define SMEM_BYTES (XS_OFF + T_SEQ*SEQ_PER_WG*4)   // 99072 bytes

// ---- fragment builders (layouts per CDNA5 ISA 7.12.2, wave32) --------------
// A (16x32 f16): lanes 0-15 row M=L hold K 0..7 / 16..23 ; lanes 16-31 row
// M=L-16 hold K 8..15 / 24..31.  KVALID<32 zero-pads the missing K columns.
template <int KVALID>
__device__ __forceinline__ v16h lds_a_frag(const _Float16* ah, int ld, int m0,
                                           int k0, int lane) {
  int row = m0 + (lane & 15);
  int kb  = (lane < 16) ? 0 : 8;
  v16h f;
#pragma unroll
  for (int g = 0; g < 2; ++g) {
#pragma unroll
    for (int j = 0; j < 8; ++j) {
      int kk = kb + g * 16 + j;
      f[g * 8 + j] = (kk < KVALID) ? ah[row * ld + k0 + kk] : (_Float16)0.0f;
    }
  }
  return f;
}

// B (32x16 f16) from row-major f32 weights W[out][in]; column N of B = row
// n0+N of W, so x @ W.T needs no transpose (f32->f16 inline; L2-resident).
template <int KVALID>
__device__ __forceinline__ v16h gbl_b_frag(const float* W, int ldw, int n0,
                                           int k0, int lane) {
  int row = n0 + (lane & 15);
  int kb  = (lane < 16) ? 0 : 8;
  v16h f;
#pragma unroll
  for (int g = 0; g < 2; ++g) {
#pragma unroll
    for (int j = 0; j < 8; ++j) {
      int kk = kb + g * 16 + j;
      f[g * 8 + j] =
          (kk < KVALID) ? (_Float16)W[row * ldw + k0 + kk] : (_Float16)0.0f;
    }
  }
  return f;
}

// C/D (16x16 f32): VGPR i holds (M = i or 8+i, N = lane&15).
__device__ __forceinline__ v8f lds_c_frag(const float* src, int ld, int m0,
                                          int lane) {
  int n  = lane & 15;
  int mb = m0 + ((lane < 16) ? 0 : 8);
  v8f c;
#pragma unroll
  for (int i = 0; i < 8; ++i) c[i] = src[(mb + i) * ld + n];
  return c;
}

// D store with per-lane pre-hoisted bias value (0.0f when no bias).
__device__ __forceinline__ void sto_d_frag(float* dst, int ld, int m0, int n0,
                                           v8f d, float bias_v, int lane) {
  int n  = lane & 15;
  int mb = m0 + ((lane < 16) ? 0 : 8);
#pragma unroll
  for (int i = 0; i < 8; ++i) dst[(mb + i) * ld + n0 + n] = d[i] + bias_v;
}

__device__ __forceinline__ v8f wmma16(v16h a, v16h b, v8f c) {
  return __builtin_amdgcn_wmma_f32_16x16x32_f16(false, a, false, b, (short)0,
                                                c, false, false);
}

// f32 row (from LDS) -> packed f16 row in A-staging buffer.
__device__ __forceinline__ void cvt4_store(_Float16* dst, const float* src) {
  float4 v = *(const float4*)src;
  h4v h;
  h[0] = (_Float16)v.x; h[1] = (_Float16)v.y;
  h[2] = (_Float16)v.z; h[3] = (_Float16)v.w;
  *(h4v*)dst = h;
}

// ---------------------------------------------------------------------------
__global__ void __launch_bounds__(NTHREADS)
temporal_transformer_fused(const float* __restrict__ x,
                           const float* __restrict__ Win,
                           const float* __restrict__ bin,
                           const float* __restrict__ Wqkv,
                           const float* __restrict__ bqkv,
                           const float* __restrict__ Wo,
                           const float* __restrict__ bo,
                           const float* __restrict__ ln1w,
                           const float* __restrict__ ln1b,
                           const float* __restrict__ W1,
                           const float* __restrict__ b1,
                           const float* __restrict__ W2,
                           const float* __restrict__ b2,
                           const float* __restrict__ ln2w,
                           const float* __restrict__ ln2b,
                           const float* __restrict__ Wout,
                           const float* __restrict__ bout,
                           float* __restrict__ out, int HW) {
  extern __shared__ char smem[];
  float*    hf = (float*)(smem + HF_OFF);     // [192][16]
  float*    cf = (float*)(smem + CF_OFF);     // [192][16]
  float*    sf = (float*)(smem + SF_OFF);     // [192][64]
  _Float16* ah = (_Float16*)(smem + AH_OFF);  // [192][64]
  float*    xs = (float*)(smem + XS_OFF);     // [24][8]

  const int tid   = threadIdx.x;
  const int lane  = tid & 31;
  const int lan16 = tid & 15;
  const int wid   = tid >> 5;
  const int nbase = blockIdx.x * SEQ_PER_WG;

  // Pull this layer's weights toward the caches (global_prefetch_b8).
  __builtin_prefetch(Wqkv + tid, 0, 0);
  __builtin_prefetch(W1 + tid, 0, 0);
  __builtin_prefetch(W2 + tid, 0, 0);

  // ---- stage input tile x[t=0..23][n=nbase..+7] into LDS via TDM -----------
#if __has_builtin(__builtin_amdgcn_tensor_load_to_lds) && \
    __has_builtin(__builtin_amdgcn_s_wait_tensorcnt)
  if (wid == 0) {
    // Tensor DMA descriptor (D#), ISA 8.3/8.4: 2D tensor [24][HW] f32,
    // tile 8 x 24 at x+nbase -> packed xs[24][8] in LDS.
    unsigned long long ga = (unsigned long long)(uintptr_t)(x + nbase);
    unsigned int       la = (unsigned int)(uintptr_t)xs;   // LDS byte offset
    v4u g0;
    g0[0] = 1u;                                         // count=1 (valid D#)
    g0[1] = la;                                         // lds_addr
    g0[2] = (unsigned int)ga;                           // global_addr[31:0]
    g0[3] = ((unsigned int)(ga >> 32) & 0x01FFFFFFu)    // global_addr[56:32]
            | (2u << 30);                               // type=2 ("image")
    v8i g1;
    g1[0] = (int)(2u << 16);                            // data_size=4B
    g1[1] = (int)(((unsigned)HW & 0xFFFFu) << 16);      // tensor_dim0[15:0]
    g1[2] = (int)(((unsigned)HW >> 16) | (24u << 16));  // dim0 hi | dim1=24
    g1[3] = (int)(8u << 16);                            // tile_dim0=8
    g1[4] = 24;                                         // tile_dim1=24
    g1[5] = HW;                                         // dim0_stride lo32
    g1[6] = 0;
    g1[7] = 0;
    v4i gz4 = {0, 0, 0, 0};
    v8i gz8 = {0, 0, 0, 0, 0, 0, 0, 0};
    __builtin_amdgcn_tensor_load_to_lds(g0, g1, gz4, gz4, gz8, 0);
    __builtin_amdgcn_s_wait_tensorcnt(0);
  }
#else
  for (int idx = tid; idx < T_SEQ * SEQ_PER_WG; idx += NTHREADS) {
    int t = idx >> 3, s = idx & 7;
    xs[idx] = x[t * HW + nbase + s];
  }
#endif
  __syncthreads();

  // ---- embed: h[r][d] = xs[t][s] * Win[d] + bin[d] -------------------------
  {
    float4 w0 = *(const float4*)(Win + 0),  w1 = *(const float4*)(Win + 4);
    float4 w2 = *(const float4*)(Win + 8),  w3 = *(const float4*)(Win + 12);
    float4 c0 = *(const float4*)(bin + 0),  c1 = *(const float4*)(bin + 4);
    float4 c2 = *(const float4*)(bin + 8),  c3 = *(const float4*)(bin + 12);
    for (int idx = tid; idx < 256; idx += NTHREADS) {   // padded 8 x 32
      int s = idx >> 5, t = idx & 31;
      if (t < T_SEQ) {
        float xv = xs[t * 8 + s];
        float* hp = hf + (s * T_SEQ + t) * 16;
        float4 r0, r1, r2, r3;
        r0.x = xv * w0.x + c0.x; r0.y = xv * w0.y + c0.y;
        r0.z = xv * w0.z + c0.z; r0.w = xv * w0.w + c0.w;
        r1.x = xv * w1.x + c1.x; r1.y = xv * w1.y + c1.y;
        r1.z = xv * w1.z + c1.z; r1.w = xv * w1.w + c1.w;
        r2.x = xv * w2.x + c2.x; r2.y = xv * w2.y + c2.y;
        r2.z = xv * w2.z + c2.z; r2.w = xv * w2.w + c2.w;
        r3.x = xv * w3.x + c3.x; r3.y = xv * w3.y + c3.y;
        r3.z = xv * w3.z + c3.z; r3.w = xv * w3.w + c3.w;
        *(float4*)(hp + 0) = r0;  *(float4*)(hp + 4) = r1;
        *(float4*)(hp + 8) = r2;  *(float4*)(hp + 12) = r3;
      }
    }
  }
  __syncthreads();

  for (int l = 0; l < NUM_LAYERS; ++l) {
    const float* Wqkv_l = Wqkv + l * 3 * D_MODEL * D_MODEL;
    const float* bqkv_l = bqkv + l * 3 * D_MODEL;
    const float* Wo_l   = Wo + l * D_MODEL * D_MODEL;
    const float* bo_l   = bo + l * D_MODEL;
    const float* W1_l   = W1 + l * DFF * D_MODEL;
    const float* b1_l   = b1 + l * DFF;
    const float* W2_l   = W2 + l * D_MODEL * DFF;
    const float* b2_l   = b2 + l * D_MODEL;
    const float* l1w = ln1w + l * D_MODEL, *l1b = ln1b + l * D_MODEL;
    const float* l2w = ln2w + l * D_MODEL, *l2b = ln2b + l * D_MODEL;

    // ---- stage f16 activations (vectorized cvt) ------------------------
    for (int i4 = tid; i4 < ROWS * 4; i4 += NTHREADS) {
      int r = i4 >> 2, c = (i4 & 3) << 2;
      cvt4_store(ah + r * 64 + c, hf + r * 16 + c);
    }
    __syncthreads();

    // ---- QKV GEMM: sf[192][48] = Ah x Wqkv^T + bqkv --------------------
    {
      v16h bq = gbl_b_frag<16>(Wqkv_l, 16, 0, 0, lane);   // B + bias hoisted
      v16h bk = gbl_b_frag<16>(Wqkv_l, 16, 16, 0, lane);  // out of the loop
      v16h bv = gbl_b_frag<16>(Wqkv_l, 16, 32, 0, lane);
      float biq = bqkv_l[lan16];
      float bik = bqkv_l[16 + lan16];
      float biv = bqkv_l[32 + lan16];
      for (int mt = wid; mt < MTILES; mt += NWAVES) {
        v16h a = lds_a_frag<16>(ah, 64, mt * 16, 0, lane);
        v8f  z = {};
        sto_d_frag(sf, 64, mt * 16, 0,  wmma16(a, bq, z), biq, lane);
        sto_d_frag(sf, 64, mt * 16, 16, wmma16(a, bk, z), bik, lane);
        sto_d_frag(sf, 64, mt * 16, 32, wmma16(a, bv, z), biv, lane);
      }
    }
    __syncthreads();

    // ---- attention: head_dim=4, T=24, online softmax in VALU -----------
    // 128 threads = 8 seq x (4 heads x 4 query-phases); 6 queries each.
    {
      int s   = tid >> 4;
      int h   = tid & 3;
      int tq0 = (tid >> 2) & 3;
      const float* base = sf + s * T_SEQ * 64;
#pragma unroll 1
      for (int i = 0; i < 6; ++i) {
        int tq = tq0 + (i << 2);
        float4 q = *(const float4*)(base + tq * 64 + h * 4);
        q.x *= 0.5f; q.y *= 0.5f; q.z *= 0.5f; q.w *= 0.5f;  // 1/sqrt(4)
        float mx = -3.4e38f, lsum = 0.f;
        float a0 = 0.f, a1 = 0.f, a2 = 0.f, a3 = 0.f;
#pragma unroll 4
        for (int tk = 0; tk < T_SEQ; ++tk) {
          float4 k4 = *(const float4*)(base + tk * 64 + 16 + h * 4);
          float4 v4 = *(const float4*)(base + tk * 64 + 32 + h * 4);
          float lg  = q.x * k4.x + q.y * k4.y + q.z * k4.z + q.w * k4.w;
          float nm  = fmaxf(mx, lg);
          float cor = __expf(mx - nm);
          float p   = __expf(lg - nm);
          lsum = lsum * cor + p;
          a0 = a0 * cor + p * v4.x;
          a1 = a1 * cor + p * v4.y;
          a2 = a2 * cor + p * v4.z;
          a3 = a3 * cor + p * v4.w;
          mx = nm;
        }
        float inv = 1.0f / lsum;
        float4 o; o.x = a0 * inv; o.y = a1 * inv; o.z = a2 * inv; o.w = a3 * inv;
        *(float4*)(cf + (s * T_SEQ + tq) * 16 + h * 4) = o;
      }
    }
    __syncthreads();

    // ---- stage f16 ctx -------------------------------------------------
    for (int i4 = tid; i4 < ROWS * 4; i4 += NTHREADS) {
      int r = i4 >> 2, c = (i4 & 3) << 2;
      cvt4_store(ah + r * 64 + c, cf + r * 16 + c);
    }
    __syncthreads();

    // ---- Wo GEMM with fused residual: hf = ctx x Wo^T + bo + hf --------
    {
      v16h b  = gbl_b_frag<16>(Wo_l, 16, 0, 0, lane);
      float bi = bo_l[lan16];
      for (int mt = wid; mt < MTILES; mt += NWAVES) {
        v16h a = lds_a_frag<16>(ah, 64, mt * 16, 0, lane);
        v8f  c = lds_c_frag(hf, 16, mt * 16, lane);   // residual rides in C
        sto_d_frag(hf, 16, mt * 16, 0, wmma16(a, b, c), bi, lane);
      }
    }
    __syncthreads();

    // ---- LayerNorm 1 (vectorized rows) ---------------------------------
    {
      float4 gw0 = *(const float4*)(l1w + 0), gw1 = *(const float4*)(l1w + 4);
      float4 gw2 = *(const float4*)(l1w + 8), gw3 = *(const float4*)(l1w + 12);
      float4 gb0 = *(const float4*)(l1b + 0), gb1 = *(const float4*)(l1b + 4);
      float4 gb2 = *(const float4*)(l1b + 8), gb3 = *(const float4*)(l1b + 12);
      for (int r = tid; r < ROWS; r += NTHREADS) {
        float* hp = hf + r * 16;
        float4 v0 = *(float4*)(hp + 0),  v1 = *(float4*)(hp + 4);
        float4 v2 = *(float4*)(hp + 8),  v3 = *(float4*)(hp + 12);
        float mu = (v0.x + v0.y + v0.z + v0.w + v1.x + v1.y + v1.z + v1.w +
                    v2.x + v2.y + v2.z + v2.w + v3.x + v3.y + v3.z + v3.w) *
                   (1.0f / 16.0f);
        float var = 0.f;
#define ACC(e) { float t_ = (e) - mu; var += t_ * t_; }
        ACC(v0.x) ACC(v0.y) ACC(v0.z) ACC(v0.w)
        ACC(v1.x) ACC(v1.y) ACC(v1.z) ACC(v1.w)
        ACC(v2.x) ACC(v2.y) ACC(v2.z) ACC(v2.w)
        ACC(v3.x) ACC(v3.y) ACC(v3.z) ACC(v3.w)
#undef ACC
        float rs = rsqrtf(var * (1.0f / 16.0f) + LN_EPS);
        v0.x = (v0.x - mu) * rs * gw0.x + gb0.x;
        v0.y = (v0.y - mu) * rs * gw0.y + gb0.y;
        v0.z = (v0.z - mu) * rs * gw0.z + gb0.z;
        v0.w = (v0.w - mu) * rs * gw0.w + gb0.w;
        v1.x = (v1.x - mu) * rs * gw1.x + gb1.x;
        v1.y = (v1.y - mu) * rs * gw1.y + gb1.y;
        v1.z = (v1.z - mu) * rs * gw1.z + gb1.z;
        v1.w = (v1.w - mu) * rs * gw1.w + gb1.w;
        v2.x = (v2.x - mu) * rs * gw2.x + gb2.x;
        v2.y = (v2.y - mu) * rs * gw2.y + gb2.y;
        v2.z = (v2.z - mu) * rs * gw2.z + gb2.z;
        v2.w = (v2.w - mu) * rs * gw2.w + gb2.w;
        v3.x = (v3.x - mu) * rs * gw3.x + gb3.x;
        v3.y = (v3.y - mu) * rs * gw3.y + gb3.y;
        v3.z = (v3.z - mu) * rs * gw3.z + gb3.z;
        v3.w = (v3.w - mu) * rs * gw3.w + gb3.w;
        *(float4*)(hp + 0) = v0;  *(float4*)(hp + 4) = v1;
        *(float4*)(hp + 8) = v2;  *(float4*)(hp + 12) = v3;
      }
    }
    __syncthreads();

    // ---- stage f16 of normalized act -----------------------------------
    for (int i4 = tid; i4 < ROWS * 4; i4 += NTHREADS) {
      int r = i4 >> 2, c = (i4 & 3) << 2;
      cvt4_store(ah + r * 64 + c, hf + r * 16 + c);
    }
    __syncthreads();

    // ---- FF up: sf[192][64] = Ah x W1^T --------------------------------
    {
      v16h b0  = gbl_b_frag<16>(W1_l, 16, 0, 0, lane);
      v16h b1f = gbl_b_frag<16>(W1_l, 16, 16, 0, lane);
      v16h b2f = gbl_b_frag<16>(W1_l, 16, 32, 0, lane);
      v16h b3f = gbl_b_frag<16>(W1_l, 16, 48, 0, lane);
      for (int mt = wid; mt < MTILES; mt += NWAVES) {
        v16h a = lds_a_frag<16>(ah, 64, mt * 16, 0, lane);
        v8f  z = {};
        sto_d_frag(sf, 64, mt * 16, 0,  wmma16(a, b0, z),  0.0f, lane);
        sto_d_frag(sf, 64, mt * 16, 16, wmma16(a, b1f, z), 0.0f, lane);
        sto_d_frag(sf, 64, mt * 16, 32, wmma16(a, b2f, z), 0.0f, lane);
        sto_d_frag(sf, 64, mt * 16, 48, wmma16(a, b3f, z), 0.0f, lane);
      }
    }
    __syncthreads();

    // ---- bias + ReLU + cvt to f16 hidden (vectorized) -------------------
    for (int i4 = tid; i4 < ROWS * 16; i4 += NTHREADS) {
      int r = i4 >> 4, c = (i4 & 15) << 2;
      float4 v = *(const float4*)(sf + r * 64 + c);
      float4 bb = *(const float4*)(b1_l + c);
      h4v h;
      h[0] = (_Float16)fmaxf(v.x + bb.x, 0.0f);
      h[1] = (_Float16)fmaxf(v.y + bb.y, 0.0f);
      h[2] = (_Float16)fmaxf(v.z + bb.z, 0.0f);
      h[3] = (_Float16)fmaxf(v.w + bb.w, 0.0f);
      *(h4v*)(ah + r * 64 + c) = h;
    }
    __syncthreads();

    // ---- FF down, K=64 (2 K-tiles), fused residual: hf += hid x W2^T ---
    {
      v16h wb0 = gbl_b_frag<32>(W2_l, 64, 0, 0, lane);
      v16h wb1 = gbl_b_frag<32>(W2_l, 64, 0, 32, lane);
      float bi = b2_l[lan16];
      for (int mt = wid; mt < MTILES; mt += NWAVES) {
        v8f  c  = lds_c_frag(hf, 16, mt * 16, lane);
        v16h a0 = lds_a_frag<32>(ah, 64, mt * 16, 0, lane);
        c = wmma16(a0, wb0, c);
        v16h a1 = lds_a_frag<32>(ah, 64, mt * 16, 32, lane);
        c = wmma16(a1, wb1, c);
        sto_d_frag(hf, 16, mt * 16, 0, c, bi, lane);
      }
    }
    __syncthreads();

    // ---- LayerNorm 2 ---------------------------------------------------
    {
      float4 gw0 = *(const float4*)(l2w + 0), gw1 = *(const float4*)(l2w + 4);
      float4 gw2 = *(const float4*)(l2w + 8), gw3 = *(const float4*)(l2w + 12);
      float4 gb0 = *(const float4*)(l2b + 0), gb1 = *(const float4*)(l2b + 4);
      float4 gb2 = *(const float4*)(l2b + 8), gb3 = *(const float4*)(l2b + 12);
      for (int r = tid; r < ROWS; r += NTHREADS) {
        float* hp = hf + r * 16;
        float4 v0 = *(float4*)(hp + 0),  v1 = *(float4*)(hp + 4);
        float4 v2 = *(float4*)(hp + 8),  v3 = *(float4*)(hp + 12);
        float mu = (v0.x + v0.y + v0.z + v0.w + v1.x + v1.y + v1.z + v1.w +
                    v2.x + v2.y + v2.z + v2.w + v3.x + v3.y + v3.z + v3.w) *
                   (1.0f / 16.0f);
        float var = 0.f;
#define ACC(e) { float t_ = (e) - mu; var += t_ * t_; }
        ACC(v0.x) ACC(v0.y) ACC(v0.z) ACC(v0.w)
        ACC(v1.x) ACC(v1.y) ACC(v1.z) ACC(v1.w)
        ACC(v2.x) ACC(v2.y) ACC(v2.z) ACC(v2.w)
        ACC(v3.x) ACC(v3.y) ACC(v3.z) ACC(v3.w)
#undef ACC
        float rs = rsqrtf(var * (1.0f / 16.0f) + LN_EPS);
        v0.x = (v0.x - mu) * rs * gw0.x + gb0.x;
        v0.y = (v0.y - mu) * rs * gw0.y + gb0.y;
        v0.z = (v0.z - mu) * rs * gw0.z + gb0.z;
        v0.w = (v0.w - mu) * rs * gw0.w + gb0.w;
        v1.x = (v1.x - mu) * rs * gw1.x + gb1.x;
        v1.y = (v1.y - mu) * rs * gw1.y + gb1.y;
        v1.z = (v1.z - mu) * rs * gw1.z + gb1.z;
        v1.w = (v1.w - mu) * rs * gw1.w + gb1.w;
        v2.x = (v2.x - mu) * rs * gw2.x + gb2.x;
        v2.y = (v2.y - mu) * rs * gw2.y + gb2.y;
        v2.z = (v2.z - mu) * rs * gw2.z + gb2.z;
        v2.w = (v2.w - mu) * rs * gw2.w + gb2.w;
        v3.x = (v3.x - mu) * rs * gw3.x + gb3.x;
        v3.y = (v3.y - mu) * rs * gw3.y + gb3.y;
        v3.z = (v3.z - mu) * rs * gw3.z + gb3.z;
        v3.w = (v3.w - mu) * rs * gw3.w + gb3.w;
        *(float4*)(hp + 0) = v0;  *(float4*)(hp + 4) = v1;
        *(float4*)(hp + 8) = v2;  *(float4*)(hp + 12) = v3;
      }
    }
    __syncthreads();
  }

  // ---- output projection: out[t, n] = h . Wout + bout ----------------------
  {
    float4 w0 = *(const float4*)(Wout + 0), w1 = *(const float4*)(Wout + 4);
    float4 w2 = *(const float4*)(Wout + 8), w3 = *(const float4*)(Wout + 12);
    float bo0 = bout[0];
    for (int idx = tid; idx < 256; idx += NTHREADS) {   // padded 8 x 32
      int s = idx >> 5, t = idx & 31;
      if (t < T_SEQ) {
        const float* hp = hf + (s * T_SEQ + t) * 16;
        float4 v0 = *(const float4*)(hp + 0),  v1 = *(const float4*)(hp + 4);
        float4 v2 = *(const float4*)(hp + 8),  v3 = *(const float4*)(hp + 12);
        float acc = bo0 +
            v0.x * w0.x + v0.y * w0.y + v0.z * w0.z + v0.w * w0.w +
            v1.x * w1.x + v1.y * w1.y + v1.z * w1.z + v1.w * w1.w +
            v2.x * w2.x + v2.y * w2.y + v2.z * w2.z + v2.w * w2.w +
            v3.x * w3.x + v3.y * w3.y + v3.z * w3.z + v3.w * w3.w;
        out[t * HW + nbase + s] = acc;
      }
    }
  }
}

// ---------------------------------------------------------------------------
extern "C" void kernel_launch(void* const* d_in, const int* in_sizes, int n_in,
                              void* d_out, int out_size, void* d_ws,
                              size_t ws_size, hipStream_t stream) {
  const float* x    = (const float*)d_in[0];
  const float* Win  = (const float*)d_in[1];
  const float* bin  = (const float*)d_in[2];
  const float* Wqkv = (const float*)d_in[3];
  const float* bqkv = (const float*)d_in[4];
  const float* Wo   = (const float*)d_in[5];
  const float* bo   = (const float*)d_in[6];
  const float* ln1w = (const float*)d_in[7];
  const float* ln1b = (const float*)d_in[8];
  const float* W1   = (const float*)d_in[9];
  const float* b1   = (const float*)d_in[10];
  const float* W2   = (const float*)d_in[11];
  const float* b2   = (const float*)d_in[12];
  const float* ln2w = (const float*)d_in[13];
  const float* ln2b = (const float*)d_in[14];
  const float* Wout = (const float*)d_in[15];
  const float* boutp = (const float*)d_in[16];
  float* out = (float*)d_out;

  int HW   = in_sizes[0] / T_SEQ;       // 65536 sequences (B=1)
  int grid = HW / SEQ_PER_WG;           // 8192 workgroups

  (void)hipFuncSetAttribute((const void*)temporal_transformer_fused,
                            hipFuncAttributeMaxDynamicSharedMemorySize,
                            SMEM_BYTES);

  temporal_transformer_fused<<<grid, NTHREADS, SMEM_BYTES, stream>>>(
      x, Win, bin, Wqkv, bqkv, Wo, bo, ln1w, ln1b, W1, b1, W2, b2, ln2w, ln2b,
      Wout, boutp, out, HW);
}
